// Patient2Vec0_51737176048065
// MI455X (gfx1250) — matrix operand order repaired and checked by reading, not compiled, not run on hardware
//
#include <hip/hip_runtime.h>
#include <hip/hip_bf16.h>

typedef __attribute__((ext_vector_type(16))) __bf16 v16bf;
typedef __attribute__((ext_vector_type(8)))  __bf16 v8bf;
typedef __attribute__((ext_vector_type(8)))  float  v8f;

// Model constants: B=128 S=64 P=32 D=256 E=256 H=128 3H=384 L=2 A=1 O=2

__device__ __forceinline__ float sigmoidf_(float x) { return 1.0f / (1.0f + __expf(-x)); }

// K index inside a 16-bit A-fragment (16x32, MxK) for element e, lane-half h.
// half 0: K = {0..7, 16..23}; half 1: +8  (per CDNA5 ISA 7.12.2)
__device__ __forceinline__ int a_kmap(int e, int half) {
    int v = e >> 1, p = e & 1;
    int base = (v < 4) ? (2 * v) : (16 + 2 * (v - 4));
    return base + p + (half ? 8 : 0);
}

// ---------------------------------------------------------------------------
// Kernel 1: fused conv scalar + weighted event sum.
//   s[b,s,p]    = dot(inputs[b,s,p,:], conv_w) + conv_b
//   conv_all[k] = sum_p s[p] * inputs[b,s,p,k]
// One block per (b,s). Memory-bound: reads the 268MB input exactly once.
// ---------------------------------------------------------------------------
__global__ __launch_bounds__(256) void conv_kernel(
    const float* __restrict__ inputs, const float* __restrict__ conv_w,
    const float* __restrict__ conv_b, float* __restrict__ conv_all)
{
    __shared__ float tile[32 * 256];
    __shared__ float sv[32];
    const int tid = threadIdx.x;
    const size_t base = (size_t)blockIdx.x * (32 * 256);

    const float4* src = (const float4*)(inputs + base);
    float4* dst4 = (float4*)tile;
#pragma unroll
    for (int i = 0; i < 8; ++i) dst4[tid + 256 * i] = src[tid + 256 * i];
    __builtin_prefetch(inputs + base + 32 * 256, 0, 1);  // global_prefetch next tile
    __syncthreads();

    // phase 1: 8 threads cooperate per p-row
    const int p = tid >> 3, g = tid & 7;
    float v = 0.0f;
    const float* row = &tile[p * 256 + g * 32];
    const float* cw  = conv_w + g * 32;
#pragma unroll
    for (int k = 0; k < 32; ++k) v += row[k] * cw[k];
    v += __shfl_down(v, 4, 8);
    v += __shfl_down(v, 2, 8);
    v += __shfl_down(v, 1, 8);
    if (g == 0) sv[p] = v + conv_b[0];
    __syncthreads();

    // phase 2: thread k accumulates over p
    float acc = 0.0f;
#pragma unroll
    for (int pp = 0; pp < 32; ++pp) acc += sv[pp] * tile[pp * 256 + tid];
    conv_all[(size_t)blockIdx.x * 256 + tid] = acc;
}

// ---------------------------------------------------------------------------
// Kernel 2: WMMA GEMM  C = act(A[M,256] @ W[N,256]^T (+bias))
//   mode 0: hardtanh clip, row-major out [M,N]        (embedding layer)
//   mode 1: +bias, out stored as [S,B,N] (xg for GRU) (input-gate precompute)
// Block = 256 threads (8 waves), tile = 128M x 16N, K=256.
// ---------------------------------------------------------------------------
__global__ __launch_bounds__(256) void gemm_bf16_kernel(
    const float* __restrict__ A, const float* __restrict__ W,
    const float* __restrict__ bias, float* __restrict__ out,
    int N, int mode)
{
    __shared__ __attribute__((aligned(16))) __bf16 Asb[128 * 264];  // padded stride
    __shared__ __attribute__((aligned(16))) __bf16 WTs[256 * 20];   // W^T [K][16] pad 20
    const int tid = threadIdx.x;
    const int bm = blockIdx.y, bn = blockIdx.x;

    const float* Ab = A + (size_t)bm * 128 * 256;
    for (int i = tid; i < 128 * 256; i += 256) {
        int r = i >> 8, k = i & 255;
        Asb[r * 264 + k] = (__bf16)Ab[i];
    }
    const float* Wb = W + (size_t)bn * 16 * 256;
    for (int i = tid; i < 16 * 256; i += 256) {
        int n = i >> 8, k = i & 255;
        WTs[k * 20 + n] = (__bf16)Wb[i];
    }
    __syncthreads();

    const int lane = tid & 31, wv = tid >> 5;
    const int half = lane >> 4, l16 = lane & 15;
    const int m0 = wv * 16;

    v8f acc = {};
#pragma unroll
    for (int kt = 0; kt < 8; ++kt) {
        v16bf a, b;
        const __bf16* arow = &Asb[(m0 + l16) * 264 + kt * 32];
#pragma unroll
        for (int e = 0; e < 16; ++e) a[e] = arow[a_kmap(e, half)];
        const __bf16* brow = &WTs[(kt * 32) * 20 + l16];
#pragma unroll
        for (int e = 0; e < 16; ++e) b[e] = brow[a_kmap(e, half) * 20];
        acc = __builtin_amdgcn_wmma_f32_16x16x32_bf16(false, a, false, b,
                                                      (short)0, acc, false, false);
    }

#pragma unroll
    for (int r = 0; r < 8; ++r) {
        int m = bm * 128 + m0 + r + 8 * half;   // D layout: VGPR r -> M = 8*(lane/16) + r
        int n = bn * 16 + l16;                  //           N = lane % 16
        float v = acc[r];
        if (mode == 0) {
            v = fminf(fmaxf(v, -1.0f), 1.0f);
            out[(size_t)m * N + n] = v;
        } else {
            v += bias[n];
            int b = m >> 6, s = m & 63;         // A rows are [B][S] flattened
            out[((size_t)s * 128 + b) * 384 + n] = v;
        }
    }
}

// ---------------------------------------------------------------------------
// Kernel 3: GRU recurrent scan, one direction per blockIdx.x, 16 batch rows
// per wave. Barrier-free: the recurrence couples only within a batch row, so
// each wave owns its rows' h state in LDS and scans all 64 steps alone.
// gh = h @ Whh^T via v_wmma_f32_16x16x32_bf16 (Whh^T staged in LDS, 96KB —
// only possible with CDNA5's 320KB-per-WGP LDS). h kept twice: bf16 row-major
// so A-frags are two ds_load_b128 per k-tile, fp32 for the z*h blend.
// ---------------------------------------------------------------------------
__global__ __launch_bounds__(128) void gru_scan_kernel(
    const float* __restrict__ xg_all,  // [2][S][B][384] (bih already added)
    const float* __restrict__ Whh2,    // [2][384][128]
    const float* __restrict__ bhh2,    // [2][384]
    float* __restrict__ y)             // [B][S][256]; dir d writes cols [128d,128d+128)
{
    __shared__ __attribute__((aligned(16))) __bf16 WTs[128 * 392]; // Whh^T [K][384] pad
    __shared__ __attribute__((aligned(16))) __bf16 hb[4 * 16 * 136]; // bf16 h (A-frags)
    __shared__ float  hf[4 * 16 * 132];                              // fp32 h (update)
    __shared__ float  bhs[384];
    const int tid = threadIdx.x;
    const int dir = blockIdx.x;
    const float* Whh = Whh2 + (size_t)dir * 384 * 128;
    const float* xg  = xg_all + (size_t)dir * 64 * 128 * 384;

    for (int i = tid; i < 384 * 128; i += 128) {
        int n = i % 384, k = i / 384;
        WTs[k * 392 + n] = (__bf16)Whh[n * 128 + k];
    }
    for (int i = tid; i < 384; i += 128) bhs[i] = bhh2[dir * 384 + i];
    for (int i = tid; i < 4 * 16 * 136; i += 128) hb[i] = (__bf16)0.0f;
    for (int i = tid; i < 4 * 16 * 132; i += 128) hf[i] = 0.0f;
    __syncthreads();

    const int lane = tid & 31, wv = tid >> 5;
    const int half = lane >> 4, l16 = lane & 15;
    const int m0 = 16 * (blockIdx.y * 4 + wv);  // batch-row base for this wave
    __bf16* hbw = &hb[wv * 16 * 136];
    float*  hfw = &hf[wv * 16 * 132];

    for (int step = 0; step < 64; ++step) {
        int t = dir ? (63 - step) : step;
        // A fragments: old h (bf16 LDS), two contiguous 16B runs per k-tile,
        // loaded before any h update this step.
        v16bf a[4];
#pragma unroll
        for (int kt = 0; kt < 4; ++kt) {
            const __bf16* hrow = &hbw[l16 * 136 + kt * 32 + half * 8];
            v8bf lo = *(const v8bf*)(hrow);        // K = kt*32 + 8*half + 0..7
            v8bf hi = *(const v8bf*)(hrow + 16);   // K = kt*32 + 16 + 8*half + 0..7
            a[kt] = __builtin_shufflevector(lo, hi, 0, 1, 2, 3, 4, 5, 6, 7,
                                            8, 9, 10, 11, 12, 13, 14, 15);
        }
        const float* xgt = xg + (size_t)t * 128 * 384;

#pragma unroll 1
        for (int tj = 0; tj < 8; ++tj) {
            v8f ar = {}, az = {}, an = {};
#pragma unroll
            for (int kt = 0; kt < 4; ++kt) {
                int krow = kt * 32;  // B frag: lane <-> N, element <-> K (a_kmap)
                const __bf16* w0 = &WTs[krow * 392 + tj * 16 + l16];
                v16bf br, bz, bn2;
#pragma unroll
                for (int e = 0; e < 16; ++e) {
                    br[e]  = w0[a_kmap(e, half) * 392];
                    bz[e]  = w0[a_kmap(e, half) * 392 + 128];
                    bn2[e] = w0[a_kmap(e, half) * 392 + 256];
                }
                ar = __builtin_amdgcn_wmma_f32_16x16x32_bf16(false, a[kt], false, br,  (short)0, ar, false, false);
                az = __builtin_amdgcn_wmma_f32_16x16x32_bf16(false, a[kt], false, bz,  (short)0, az, false, false);
                an = __builtin_amdgcn_wmma_f32_16x16x32_bf16(false, a[kt], false, bn2, (short)0, an, false, false);
            }
            int j = tj * 16 + l16;
            float bhr = bhs[j], bhz = bhs[128 + j], bhn = bhs[256 + j];
#pragma unroll
            for (int r = 0; r < 8; ++r) {
                int mloc = r + 8 * half;
                int brow = m0 + mloc;
                const float* xr = &xgt[(size_t)brow * 384];
                float rg = sigmoidf_(xr[j]       + ar[r] + bhr);
                float zg = sigmoidf_(xr[128 + j] + az[r] + bhz);
                float ng = tanhf(xr[256 + j] + rg * (an[r] + bhn));
                float hold = hfw[mloc * 132 + j];
                float hnew = (1.0f - zg) * ng + zg * hold;
                hfw[mloc * 132 + j] = hnew;
                hbw[mloc * 136 + j] = (__bf16)hnew;
                y[((size_t)brow * 64 + t) * 256 + dir * 128 + j] = hnew;
            }
        }
        // no barrier: this wave reads/writes only its own h rows; a[] holds old h
    }
}

// ---------------------------------------------------------------------------
// Kernel 4: attention (A=1 => softmax over one logit => alpha == 1,
// context = sum_s states) + linear + softmax output head.
// ---------------------------------------------------------------------------
__global__ __launch_bounds__(256) void att_out_kernel(
    const float* __restrict__ lin_w, const float* __restrict__ lin_b,
    float* __restrict__ dout)
{
    __shared__ float ctx[256];
    __shared__ float lg[2];
    const int b = blockIdx.x, tid = threadIdx.x;
    const float* states = dout + 256;                       // [B,S,256]
    float* context = dout + 256 + 128 * 64 * 256;           // [B,256]
    float* alpha   = context + 128 * 256;                   // [B,1,S]

    float c = 0.0f;
    for (int s = 0; s < 64; ++s) c += states[((size_t)b * 64 + s) * 256 + tid];
    context[b * 256 + tid] = c;
    ctx[tid] = c;
    if (tid < 64) alpha[b * 64 + tid] = 1.0f;
    __syncthreads();
    if (tid < 2) {
        float l = lin_b[tid];
        for (int h = 0; h < 256; ++h) l += ctx[h] * lin_w[tid * 256 + h];
        lg[tid] = l;
    }
    __syncthreads();
    if (tid == 0) {
        float m = fmaxf(lg[0], lg[1]);
        float e0 = __expf(lg[0] - m), e1 = __expf(lg[1] - m);
        float inv = 1.0f / (e0 + e1);
        dout[b * 2 + 0] = e0 * inv;
        dout[b * 2 + 1] = e1 * inv;
    }
}

// ---------------------------------------------------------------------------
extern "C" void kernel_launch(void* const* d_in, const int* in_sizes, int n_in,
                              void* d_out, int out_size, void* d_ws, size_t ws_size,
                              hipStream_t stream)
{
    (void)in_sizes; (void)n_in; (void)out_size; (void)ws_size;
    const float* inputs  = (const float*)d_in[0];
    const float* conv_w  = (const float*)d_in[1];
    const float* conv_b  = (const float*)d_in[2];
    const float* embed_w = (const float*)d_in[3];
    const float* gWih    = (const float*)d_in[4];   // [2][2][384][256]
    const float* gWhh    = (const float*)d_in[5];   // [2][2][384][128]
    const float* gbih    = (const float*)d_in[6];   // [2][2][384]
    const float* gbhh    = (const float*)d_in[7];   // [2][2][384]
    const float* lin_w   = (const float*)d_in[9];
    const float* lin_b   = (const float*)d_in[10];

    float* ws = (float*)d_ws;
    float* conv_all = ws;              // 2,097,152 floats (reused as layer-0 output y0)
    float* x0 = ws + 2097152;          // 2,097,152 floats
    float* xg = ws + 4194304;          // 2 * 64*128*384 = 6,291,456 floats
    float* dout = (float*)d_out;
    float* states = dout + 256;        // [B,S,256] final GRU layer output

    // 1. conv + event-weighted sum  -> conv_all [B*S, 256]
    conv_kernel<<<8192, 256, 0, stream>>>(inputs, conv_w, conv_b, conv_all);
    // 2. embedding: clip(conv_all @ embed_w^T) -> x0 [B*S, 256]
    gemm_bf16_kernel<<<dim3(16, 64), 256, 0, stream>>>(conv_all, embed_w, nullptr, x0, 256, 0);

    // 3. layer 0: xg = x0 @ Wih^T + bih (both dirs), then scan -> y0 (reuses conv_all)
    for (int d = 0; d < 2; ++d)
        gemm_bf16_kernel<<<dim3(24, 64), 256, 0, stream>>>(
            x0, gWih + (size_t)(0 * 2 + d) * 384 * 256, gbih + (0 * 2 + d) * 384,
            xg + (size_t)d * 64 * 128 * 384, 384, 1);
    gru_scan_kernel<<<dim3(2, 2), 128, 0, stream>>>(xg, gWhh, gbhh, conv_all);

    // 4. layer 1: xg = y0 @ Wih^T + bih, scan -> states (d_out)
    for (int d = 0; d < 2; ++d)
        gemm_bf16_kernel<<<dim3(24, 64), 256, 0, stream>>>(
            conv_all, gWih + (size_t)(1 * 2 + d) * 384 * 256, gbih + (1 * 2 + d) * 384,
            xg + (size_t)d * 64 * 128 * 384, 384, 1);
    gru_scan_kernel<<<dim3(2, 2), 128, 0, stream>>>(
        xg, gWhh + 2 * 384 * 128, gbhh + 2 * 384, states);

    // 5. attention (alpha==1) + context + output softmax
    att_out_kernel<<<128, 256, 0, stream>>>(lin_w, lin_b, dout);
}
